// GCN_35210141893299
// MI455X (gfx1250) — matrix operand order repaired and checked by reading, not compiled, not run on hardware
//
#include <hip/hip_runtime.h>
#include <hip/hip_bf16.h>

// ---------------------------------------------------------------------------
// 2-layer GCN inference for MI455X (gfx1250, wave32).
//
// Cost model: GEMMs are ~5 GFLOP total (noise on the WMMA units). The edge
// aggregation moves ~2.6 GB minimum (gather + atomic scatter) -> ~110us at
// 23.3 TB/s HBM; the per-layer working set (~130 MB) sits in the 192 MB L2,
// so global_atomic_add_f32 resolves in L2. Structure:
//   - bf16 WMMA GEMM with W staged per-block in LDS in fragment order
//     (2x ds_load_b128 per B fragment instead of 16 strided global loads)
//   - wave-per-edge float4 gather + f32 atomic scatter for aggregation.
// ---------------------------------------------------------------------------

typedef __bf16 v16bf __attribute__((ext_vector_type(16)));
typedef float  v8f   __attribute__((ext_vector_type(8)));

#define N_NODES 100000

// ---------------- degree / norm ----------------
__global__ void init_deg(int* deg, int n) {
  int i = blockIdx.x * blockDim.x + threadIdx.x;
  if (i < n) deg[i] = 0;
}

__global__ void count_deg(const long long* __restrict__ ei, int E, int* __restrict__ deg) {
  int e = blockIdx.x * blockDim.x + threadIdx.x;
  if (e < E) atomicAdd(&deg[(int)ei[(size_t)E + e]], 1);   // dst row of edge_index
}

__global__ void make_dinv(const int* __restrict__ deg, float* __restrict__ dinv, int n) {
  int i = blockIdx.x * blockDim.x + threadIdx.x;
  if (i < n) dinv[i] = __frsqrt_rn((float)(deg[i] + 1));   // +1 self-loop; always > 0
}

// ---------------- WMMA GEMM: Y[M,N] = (relu?)X[M,128] @ W[128,N] ----------------
// One wave computes a 16-row strip across all NT 16-wide column tiles.
// Fragment layouts per CDNA5 ISA sec 7.12.2:
//   A (16-bit, 16x32): lane half h, row m=lane&15:
//       a[j]   = X[m][kb + 8h + j],  a[8+j] = X[m][kb + 16 + 8h + j]   j=0..7
//   B (16-bit, 32x16): VGPR g<4 holds K=2g,2g+1, g>=4 holds K=2g+8,2g+9; lanes 16-31 at K+8:
//       b[j]   = W[kb + 8h + j][n0 + (lane&15)],  b[8+j] = W[kb + 16 + 8h + j][n0 + (lane&15)]   j=0..7
//   C/D (f32, 16x16):  acc[v] -> D[m0 + 8h + v][n0 + (lane&15)]
// W is staged once per block into LDS, pre-swizzled so each (fragment,lane)
// pair is 32 contiguous bytes -> inner loop reads are 2x ds_load_b128.
template <int NT, bool RELU>
__global__ void gemm_wmma_bf16(const float* __restrict__ X,
                               const float* __restrict__ W,
                               float* __restrict__ Y, int M) {
  constexpr int K = 128;
  constexpr int N = NT * 16;
  constexpr int NFRAG = (K / 32) * NT;            // fragments over (kb, tile)
  __shared__ v16bf Bfrag[NFRAG * 32];             // NT=8: 32 KB, NT=4: 16 KB

  const int tid  = (int)threadIdx.x;
  const int lane = tid & 31;
  const int half = lane >> 4;
  const int lrow = lane & 15;

  // ---- cooperative staging of W -> LDS in fragment order (global-coalesced)
  {
    __bf16* Wl = (__bf16*)Bfrag;
    for (int e = tid; e < K * N; e += (int)blockDim.x) {
      int n = e & (N - 1);                        // consecutive threads: consecutive n
      int k = e >> (NT == 8 ? 7 : 6);
      int kbi = k >> 5, r = k & 31;
      int h = (r >> 3) & 1, j = (r & 7) + ((r >> 4) << 3);
      int t = n >> 4,  lr = n & 15;
      Wl[((kbi * NT + t) << 9) + ((h * 16 + lr) << 4) + j] = (__bf16)W[e];
    }
  }
  __syncthreads();                                // all waves participate

  const int wave = (int)((blockIdx.x * blockDim.x + threadIdx.x) >> 5);
  const int m0   = wave * 16;
  if (m0 >= M) return;                            // wave-uniform: EXEC all-1s for WMMA

  v8f acc[NT];
#pragma unroll
  for (int t = 0; t < NT; ++t)
#pragma unroll
    for (int j = 0; j < 8; ++j) acc[t][j] = 0.0f;

  const float4* xrow4 = (const float4*)(X + (size_t)(m0 + lrow) * K);

#pragma unroll
  for (int kb = 0; kb < K; kb += 32) {
    // A fragment: two 8-float runs -> 4x global_load_b128
    float4 p0 = xrow4[(kb + 8 * half) >> 2];
    float4 p1 = xrow4[((kb + 8 * half) >> 2) + 1];
    float4 q0 = xrow4[(kb + 16 + 8 * half) >> 2];
    float4 q1 = xrow4[((kb + 16 + 8 * half) >> 2) + 1];
    if (RELU) {
      p0.x = fmaxf(p0.x, 0.f); p0.y = fmaxf(p0.y, 0.f); p0.z = fmaxf(p0.z, 0.f); p0.w = fmaxf(p0.w, 0.f);
      p1.x = fmaxf(p1.x, 0.f); p1.y = fmaxf(p1.y, 0.f); p1.z = fmaxf(p1.z, 0.f); p1.w = fmaxf(p1.w, 0.f);
      q0.x = fmaxf(q0.x, 0.f); q0.y = fmaxf(q0.y, 0.f); q0.z = fmaxf(q0.z, 0.f); q0.w = fmaxf(q0.w, 0.f);
      q1.x = fmaxf(q1.x, 0.f); q1.y = fmaxf(q1.y, 0.f); q1.z = fmaxf(q1.z, 0.f); q1.w = fmaxf(q1.w, 0.f);
    }
    v16bf a;
    a[0]  = (__bf16)p0.x; a[1]  = (__bf16)p0.y; a[2]  = (__bf16)p0.z; a[3]  = (__bf16)p0.w;
    a[4]  = (__bf16)p1.x; a[5]  = (__bf16)p1.y; a[6]  = (__bf16)p1.z; a[7]  = (__bf16)p1.w;
    a[8]  = (__bf16)q0.x; a[9]  = (__bf16)q0.y; a[10] = (__bf16)q0.z; a[11] = (__bf16)q0.w;
    a[12] = (__bf16)q1.x; a[13] = (__bf16)q1.y; a[14] = (__bf16)q1.z; a[15] = (__bf16)q1.w;

#pragma unroll
    for (int t = 0; t < NT; ++t) {
      v16bf b = Bfrag[((kb >> 5) * NT + t) * 32 + lane];   // 2x ds_load_b128
      acc[t] = __builtin_amdgcn_wmma_f32_16x16x32_bf16(
          /*neg_a=*/false, a, /*neg_b=*/false, b,
          /*c_mod=*/(short)0, acc[t], /*reuse_a=*/false, /*reuse_b=*/false);
    }
  }

#pragma unroll
  for (int t = 0; t < NT; ++t) {
    const int n0 = t * 16;
#pragma unroll
    for (int v = 0; v < 8; ++v)
      Y[(size_t)(m0 + 8 * half + v) * N + (n0 + lrow)] = acc[t][v];
  }
}

// ---------------- out[i,:] = bias + dinv[i]^2 * xw[i,:]  (self-loop + bias, no atomics)
template <int F>
__global__ void self_init(const float* __restrict__ xw, const float* __restrict__ dinv,
                          const float* __restrict__ bias, float* __restrict__ out, int n) {
  int i = blockIdx.x * blockDim.x + threadIdx.x;
  if (i >= n * F) return;
  int node = i >> (F == 128 ? 7 : 6);
  int f    = i & (F - 1);
  float dv = dinv[node];
  out[i] = bias[f] + dv * dv * xw[i];
}

// ---------------- one wave per edge: out[dst,:] += dinv[s]*dinv[d]*xw[src,:]
template <int F>
__global__ void edge_scatter(const long long* __restrict__ ei,
                             const float* __restrict__ xw,
                             const float* __restrict__ dinv,
                             float* __restrict__ out, int E) {
  unsigned gid = blockIdx.x * blockDim.x + threadIdx.x;
  int e    = (int)(gid >> 5);
  int lane = (int)(gid & 31u);
  if (e >= E) return;
  int s = (int)ei[e];
  int d = (int)ei[(size_t)E + e];
  float w = dinv[s] * dinv[d];
  const float* srow = xw + (size_t)s * F;
  float* drow = out + (size_t)d * F;
  if constexpr (F == 128) {
    const float4 v = ((const float4*)srow)[lane];   // 32 lanes * 16B = full 512B row
    atomicAdd(drow + lane * 4 + 0, v.x * w);
    atomicAdd(drow + lane * 4 + 1, v.y * w);
    atomicAdd(drow + lane * 4 + 2, v.z * w);
    atomicAdd(drow + lane * 4 + 3, v.w * w);
  } else {                                          // F == 64
    const float2 v = ((const float2*)srow)[lane];
    atomicAdd(drow + lane * 2 + 0, v.x * w);
    atomicAdd(drow + lane * 2 + 1, v.y * w);
  }
}

// ---------------------------------------------------------------------------
extern "C" void kernel_launch(void* const* d_in, const int* in_sizes, int n_in,
                              void* d_out, int out_size, void* d_ws, size_t ws_size,
                              hipStream_t stream) {
  const float*     x  = (const float*)d_in[0];
  const long long* ei = (const long long*)d_in[1];   // int64 edge_index [2,E]
  const float*     W1 = (const float*)d_in[2];
  const float*     b1 = (const float*)d_in[3];
  const float*     W2 = (const float*)d_in[4];
  const float*     b2 = (const float*)d_in[5];

  const int Nn = N_NODES;
  const int E  = in_sizes[1] / 2;

  // workspace layout (offsets 256B-aligned): deg | dinv | xw (reused) | h
  char*  ws   = (char*)d_ws;
  int*   deg  = (int*)(ws + 0);                          // 400 KB
  float* dinv = (float*)(ws + 409600);                   // 400 KB
  float* xw   = (float*)(ws + (1u << 20));               // 51.2 MB (xw1, then xw2)
  float* h    = (float*)(ws + (1u << 20) + 51200000u);   // 51.2 MB
  float* out  = (float*)d_out;

  const int T = 256;

  // ---- symmetric normalization terms
  init_deg<<<(Nn + T - 1) / T, T, 0, stream>>>(deg, Nn);
  count_deg<<<(E + T - 1) / T, T, 0, stream>>>(ei, E, deg);
  make_dinv<<<(Nn + T - 1) / T, T, 0, stream>>>(deg, dinv, Nn);

  const int waves   = Nn / 16;                       // 6250 row tiles
  const int gemmBlk = (waves * 32 + T - 1) / T;
  const int edgeBlk = (int)(((long long)E * 32 + T - 1) / T);

  // ---- layer 1: h = b1 + scatter(dinv-normed xw1)
  gemm_wmma_bf16<8, false><<<gemmBlk, T, 0, stream>>>(x, W1, xw, Nn);
  self_init<128><<<(Nn * 128 + T - 1) / T, T, 0, stream>>>(xw, dinv, b1, h, Nn);
  edge_scatter<128><<<edgeBlk, T, 0, stream>>>(ei, xw, dinv, h, E);

  // ---- layer 2: out = b2 + scatter(dinv-normed relu(h)@W2); ReLU fused in A load
  gemm_wmma_bf16<4, true><<<gemmBlk, T, 0, stream>>>(h, W2, xw, Nn);
  self_init<64><<<(Nn * 64 + T - 1) / T, T, 0, stream>>>(xw, dinv, b2, out, Nn);
  edge_scatter<64><<<edgeBlk, T, 0, stream>>>(ei, xw, dinv, out, E);
}
